// GaussianCodebook_9680856285671
// MI455X (gfx1250) — matrix operand order repaired and checked by reading, not compiled, not run on hardware
//
#include <hip/hip_runtime.h>
#include <hip/hip_bf16.h>
#include <stdint.h>

// Problem constants (from reference): B=8, S=4096, D=512, C=2048
static constexpr int Mdim = 8 * 4096;   // 32768 rows (b,s flattened)
static constexpr int Kdim = 512;        // D
static constexpr int Ndim = 2048;       // C

typedef __attribute__((ext_vector_type(16))) __bf16 v16bf;
typedef __attribute__((ext_vector_type(8)))  float  v8f;

union Frag {
  v16bf v;
  uint4 q[2];
};

__device__ __forceinline__ unsigned short f2bf(float f) {
  // round-to-nearest-even f32 -> bf16 (inputs are finite; NaN path not needed)
  unsigned int u = __float_as_uint(f);
  u += 0x7FFFu + ((u >> 16) & 1u);
  return (unsigned short)(u >> 16);
}

// ---------------------------------------------------------------------------
// x [M,K] f32 -> xbf [M,K] bf16, plus x2[m] = mean_k x^2 (fused wave reduce)
// one wave per row, 8 rows per 256-thread block
// ---------------------------------------------------------------------------
__global__ __launch_bounds__(256) void gc_convert_x(
    const float* __restrict__ x, unsigned short* __restrict__ xbf,
    float* __restrict__ x2) {
  const int lane = threadIdx.x & 31;
  const int row  = blockIdx.x * 8 + (threadIdx.x >> 5);
  const float* src = x + (size_t)row * Kdim;
  unsigned short* dst = xbf + (size_t)row * Kdim;
  float ss = 0.f;
#pragma unroll
  for (int i = 0; i < 4; ++i) {
    int off = i * 128 + lane * 4;            // coalesced float4 per lane
    float4 v = *(const float4*)(src + off);
    ss += v.x * v.x + v.y * v.y + v.z * v.z + v.w * v.w;
    ushort4 b;
    b.x = f2bf(v.x); b.y = f2bf(v.y); b.z = f2bf(v.z); b.w = f2bf(v.w);
    *(ushort4*)(dst + off) = b;
  }
#pragma unroll
  for (int off = 16; off > 0; off >>= 1) ss += __shfl_xor(ss, off, 32);
  if (lane == 0) x2[row] = ss * (1.0f / (float)Kdim);
}

// ---------------------------------------------------------------------------
// cb [K,N] f32 -> cbT [N,K] bf16 via LDS 64x64 tile transpose (coalesced both ways)
// ---------------------------------------------------------------------------
__global__ __launch_bounds__(256) void gc_transpose_cb(
    const float* __restrict__ cb, unsigned short* __restrict__ cbT) {
  __shared__ unsigned short t[64 * 68];      // padded to dodge bank conflicts
  const int n0 = blockIdx.x * 64;
  const int k0 = blockIdx.y * 64;
#pragma unroll
  for (int p = 0; p < 16; ++p) {
    int e = threadIdx.x + p * 256;           // 4096 elems / 256 threads
    int k = e >> 6, n = e & 63;              // consecutive tid -> consecutive n (coalesced)
    t[n * 68 + k] = f2bf(cb[(size_t)(k0 + k) * Ndim + n0 + n]);
  }
  __syncthreads();
#pragma unroll
  for (int p = 0; p < 16; ++p) {
    int e = threadIdx.x + p * 256;
    int n = e >> 6, k = e & 63;              // consecutive tid -> consecutive k (coalesced)
    cbT[(size_t)(n0 + n) * Kdim + k0 + k] = t[n * 68 + k];
  }
}

// ---------------------------------------------------------------------------
// c2[n] = mean_k cb[k][n]^2  (threads span n -> coalesced row reads)
// ---------------------------------------------------------------------------
__global__ __launch_bounds__(256) void gc_c2(
    const float* __restrict__ cb, float* __restrict__ c2) {
  int n = blockIdx.x * 256 + threadIdx.x;
  float s = 0.f;
  for (int k = 0; k < Kdim; ++k) {
    float v = cb[(size_t)k * Ndim + n];
    s += v * v;
  }
  c2[n] = s * (1.0f / (float)Kdim);
}

// ---------------------------------------------------------------------------
// GEMM: out[m][n] = x2[m] + c2[n] - (2/K) * sum_k xbf[m][k]*cbT[n][k]
// 128x128 block tile, 8 waves, each wave = 2x4 grid of 16x16 WMMA tiles.
// Double-buffered LDS fed by GLOBAL_LOAD_ASYNC_TO_LDS_B128 in GVS mode
// (SGPR base + i32 VGPR offset, ASYNCcnt-tracked). The i32 offset is advanced
// *inside* the asm ("+v" operand) so the compiler keeps one loop-carried
// register per stream instead of materializing all 16 k-step addresses.
// bf16 operands, f32 accumulation via v_wmma_f32_16x16x32_bf16.
// ---------------------------------------------------------------------------
__global__ __launch_bounds__(256) void gc_gemm(
    const unsigned short* __restrict__ xbf,
    const unsigned short* __restrict__ cbT,
    const float* __restrict__ x2,
    const float* __restrict__ c2,
    float* __restrict__ out) {
  constexpr int LDT = 40;                    // ushorts per LDS row: 32 data + 8 pad (80B, 16B-aligned)
  __shared__ unsigned short sA[2][128 * LDT];
  __shared__ unsigned short sB[2][128 * LDT];

  const int tid  = threadIdx.x;
  const int lane = tid & 31;
  const int w    = tid >> 5;
  const int wrow = w & 3;                    // m offset: wrow*32
  const int wcol = w >> 2;                   // n offset: wcol*64
  const int m0   = blockIdx.y * 128;
  const int n0   = blockIdx.x * 128;

  v8f acc[2][4];
  {
    v8f zero = {};
#pragma unroll
    for (int i = 0; i < 2; ++i)
#pragma unroll
      for (int j = 0; j < 4; ++j) acc[i][j] = zero;
  }

  // Per-thread copy-engine state: i32 byte offsets into xbf/cbT (GVS voffset),
  // and the (buffer-dependent) LDS destination byte addresses.
  unsigned voffA[2], voffB[2];
  unsigned ldsA[2][2], ldsB[2][2];           // [p][buf]
#pragma unroll
  for (int p = 0; p < 2; ++p) {
    int chunk = tid + p * 256;               // 512 chunks of 16B per matrix tile
    int row   = chunk >> 2;                  // 0..127
    int off   = (chunk & 3) * 8;             // ushort offset within row
    voffA[p] = (unsigned)(((m0 + row) * Kdim + off) * 2);   // bytes
    voffB[p] = (unsigned)(((n0 + row) * Kdim + off) * 2);   // bytes
#pragma unroll
    for (int b = 0; b < 2; ++b) {
      // Generic LDS pointer low 32 bits == LDS byte address (aperture mapping).
      ldsA[p][b] = (unsigned)(size_t)&sA[b][row * LDT + off];
      ldsB[p][b] = (unsigned)(size_t)&sB[b][row * LDT + off];
    }
  }

  // Issue async copies for one 32-deep k-slice into LDS buffer `buf`; the
  // voffset register is advanced by 64 bytes inside the asm (opaque to the
  // compiler -> stays loop-carried). Overwriting the address VGPR right after
  // issue is hazard-free per ISA (addresses consumed at issue).
  auto issue_tiles = [&](int buf) {
#pragma unroll
    for (int p = 0; p < 2; ++p) {
      asm volatile("global_load_async_to_lds_b128 %1, %0, %2\n\t"
                   "v_add_nc_u32_e32 %0, 64, %0"
                   : "+v"(voffA[p])
                   : "v"(ldsA[p][buf]), "s"(xbf)
                   : "memory");
      asm volatile("global_load_async_to_lds_b128 %1, %0, %2\n\t"
                   "v_add_nc_u32_e32 %0, 64, %0"
                   : "+v"(voffB[p])
                   : "v"(ldsB[p][buf]), "s"(cbT)
                   : "memory");
    }
  };

#define ASYNC_WAIT() asm volatile("s_wait_asynccnt 0x0" ::: "memory")

  issue_tiles(0);

  // ISA 16-bit A fragment: lane holds row (lane&15), K in [kbA,kbA+8) U [kbA+16,kbA+24)
  const int kbA  = (lane >> 4) * 8;
  // ISA 16-bit B fragment: lane holds col (lane&15), K in [kbB,kbB+8) U [kbB+16,kbB+24)
  const int kbB  = (lane >> 4) * 8;
  const int rsel = lane & 15;

  constexpr int KT = Kdim / 32;              // 16 k-steps
  for (int kt = 0; kt < KT; ++kt) {
    ASYNC_WAIT();                            // this wave's loads into buf kt&1 have landed
    __syncthreads();                         // ...and everyone else's too; prior reads of other buf done
    if (kt + 1 < KT) issue_tiles((kt + 1) & 1);
    const int buf = kt & 1;

    Frag fa[2];
    Frag fb[4];
#pragma unroll
    for (int mt = 0; mt < 2; ++mt) {
      const unsigned short* pa = &sA[buf][(wrow * 32 + mt * 16 + rsel) * LDT + kbA];
      fa[mt].q[0] = *(const uint4*)(pa);
      fa[mt].q[1] = *(const uint4*)(pa + 16);
    }
#pragma unroll
    for (int nt = 0; nt < 4; ++nt) {
      const unsigned short* pb = &sB[buf][(wcol * 64 + nt * 16 + rsel) * LDT + kbB];
      fb[nt].q[0] = *(const uint4*)(pb);
      fb[nt].q[1] = *(const uint4*)(pb + 16);
    }
#pragma unroll
    for (int mt = 0; mt < 2; ++mt)
#pragma unroll
      for (int nt = 0; nt < 4; ++nt)
        acc[mt][nt] = __builtin_amdgcn_wmma_f32_16x16x32_bf16(
            false, fa[mt].v, false, fb[nt].v, (short)0, acc[mt][nt],
            false, false);
  }

  // Epilogue: fuse distance terms. C/D layout: VGPR j, lane l -> M = j + 8*(l>>4), N = l&15
  // Output is a 256MB write-once stream (> L2): use non-temporal stores so it
  // doesn't evict the L2-resident bf16 x / codebook working set.
  const float scale = 2.0f / (float)Kdim;
#pragma unroll
  for (int mt = 0; mt < 2; ++mt) {
#pragma unroll
    for (int nt = 0; nt < 4; ++nt) {
      int n = n0 + wcol * 64 + nt * 16 + (lane & 15);
      float c2n = c2[n];
#pragma unroll
      for (int j = 0; j < 8; ++j) {
        int m = m0 + wrow * 32 + mt * 16 + (lane >> 4) * 8 + j;
        float val = x2[m] + c2n - scale * acc[mt][nt][j];
        __builtin_nontemporal_store(val, &out[(size_t)m * Ndim + n]);
      }
    }
  }
}

// ---------------------------------------------------------------------------
extern "C" void kernel_launch(void* const* d_in, const int* in_sizes, int n_in,
                              void* d_out, int out_size, void* d_ws, size_t ws_size,
                              hipStream_t stream) {
  const float* x  = (const float*)d_in[0];   // [8,4096,512] f32
  const float* cb = (const float*)d_in[1];   // [1,1,512,2048] f32
  float* out = (float*)d_out;                // [8,4096,2048] f32

  // Workspace layout (~34.2 MB):
  unsigned short* xbf = (unsigned short*)d_ws;                 // 32768*512 bf16 (32 MB)
  unsigned short* cbT = xbf + (size_t)Mdim * Kdim;             // 2048*512 bf16 (2 MB)
  float* x2 = (float*)(cbT + (size_t)Ndim * Kdim);             // 32768 f32
  float* c2 = x2 + Mdim;                                       // 2048 f32

  gc_convert_x  <<<Mdim / 8, 256, 0, stream>>>(x, xbf, x2);
  gc_transpose_cb<<<dim3(Ndim / 64, Kdim / 64), 256, 0, stream>>>(cb, cbT);
  gc_c2         <<<Ndim / 256, 256, 0, stream>>>(cb, c2);
  gc_gemm       <<<dim3(Ndim / 128, Mdim / 128), 256, 0, stream>>>(xbf, cbT, x2, c2, out);
}
